// GraphConvLayer_67705864454404
// MI455X (gfx1250) — matrix operand order repaired and checked by reading, not compiled, not run on hardware
//
#include <hip/hip_runtime.h>
#include <hip/hip_bf16.h>

typedef __bf16 bf16_t;
typedef bf16_t v8bf  __attribute__((ext_vector_type(8)));
typedef bf16_t v16bf __attribute__((ext_vector_type(16)));
typedef float  v8f   __attribute__((ext_vector_type(8)));

#define LOG2E 1.44269504088896340736f

constexpr int Bn = 4, Nn = 8192, Dn = 64;
constexpr int QT = 128;   // query rows per block (8 waves * 16)
constexpr int KT = 64;    // key rows per tile

// ---- prep: fp32 -> bf16 row-major, bf16 transposed, kernel transposed ----
__global__ void prep_kernel(const float* __restrict__ X, const float* __restrict__ W,
                            bf16_t* __restrict__ Xb, bf16_t* __restrict__ XbT,
                            bf16_t* __restrict__ Wt)
{
    int i = blockIdx.x * blockDim.x + threadIdx.x;
    const int total = Bn * Nn * Dn;
    if (i < total) {
        int d = i & (Dn - 1);
        int n = (i / Dn) & (Nn - 1);
        int b = i / (Dn * Nn);
        bf16_t v = (bf16_t)X[i];
        Xb[i] = v;
        XbT[((long)b * Dn + d) * Nn + n] = v;
    }
    if (i < Dn * 64) {
        int dd = i >> 6, oo = i & 63;
        Wt[oo * Dn + dd] = (bf16_t)W[dd * 64 + oo];
    }
}

__device__ __forceinline__ v16bf make_afrag(const bf16_t* p) {
    // A-layout 16x32 bf16: this lane reads K = off..off+15 (contiguous)
    v8bf a = *(const v8bf*)(p);
    v8bf c = *(const v8bf*)(p + 8);
    v16bf f;
#pragma unroll
    for (int j = 0; j < 8; ++j) { f[j] = a[j]; f[j + 8] = c[j]; }
    return f;
}

__global__ __launch_bounds__(256, 1)
void flashattn_kernel(const bf16_t* __restrict__ Xb, const bf16_t* __restrict__ XbT,
                      const bf16_t* __restrict__ Wt, float* __restrict__ Out)
{
    __shared__ bf16_t lds[8][16 * Dn];   // per-wave 16x64 bf16 staging slab (16 KB)

    const int wave = threadIdx.x >> 5;
    const int lane = threadIdx.x & 31;
    const int lo   = lane & 15;
    const int hi   = lane >> 4;

    const int b  = blockIdx.y;
    const int q0 = blockIdx.x * QT + wave * 16;

    const bf16_t* Xbb  = Xb  + (long)b * Nn * Dn;
    const bf16_t* XbTb = XbT + (long)b * Dn * Nn;

    // ---- Q fragments (A-layout, kept in registers for the whole sweep) ----
    v16bf qf[2];
    {
        const bf16_t* qrow = Xbb + (long)(q0 + lo) * Dn;
#pragma unroll
        for (int kc = 0; kc < 2; ++kc)
            qf[kc] = make_afrag(qrow + kc * 32 + hi * 16);
    }

    v8f o[4];
    float mrow[8], lrow[8];
#pragma unroll
    for (int t = 0; t < 4; ++t) { v8f z = {}; o[t] = z; }
#pragma unroll
    for (int r = 0; r < 8; ++r) { mrow[r] = -1e30f; lrow[r] = 0.0f; }

    bf16_t* slab = &lds[wave][0];

    for (int k0 = 0; k0 < Nn; k0 += KT) {
        if (k0 + KT < Nn) {
            __builtin_prefetch(Xbb + (long)(k0 + KT + lo) * Dn, 0, 1);
            __builtin_prefetch(XbTb + (long)lo * Nn + (k0 + KT), 0, 1);
        }
        // ---- S = Q * K^T : 16x64 (4 n-tiles of 16 keys) ----
        v8f s[4];
#pragma unroll
        for (int t = 0; t < 4; ++t) {
            v8f acc = {};
#pragma unroll
            for (int kc = 0; kc < 2; ++kc) {
                // B-layout 32x16: lane holds key column k0+t*16+lo, contiguous 16 d-values
                const bf16_t* krow = Xbb + (long)(k0 + t * 16 + lo) * Dn + kc * 32 + hi * 16;
                v16bf kb = *(const v16bf*)krow;
                acc = __builtin_amdgcn_wmma_f32_16x16x32_bf16(false, qf[kc], false, kb,
                                                              (short)0, acc, false, false);
            }
            s[t] = acc;
        }
        // ---- online softmax (rows live in 16-lane halves) ----
#pragma unroll
        for (int r = 0; r < 8; ++r) {
            float tm = s[0][r];
            tm = fmaxf(tm, s[1][r]); tm = fmaxf(tm, s[2][r]); tm = fmaxf(tm, s[3][r]);
#pragma unroll
            for (int msk = 1; msk <= 8; msk <<= 1)
                tm = fmaxf(tm, __shfl_xor(tm, msk, 32));
            float mnew  = fmaxf(mrow[r], tm);
            float scale = __builtin_amdgcn_exp2f((mrow[r] - mnew) * LOG2E);
            float ps = 0.0f;
#pragma unroll
            for (int t = 0; t < 4; ++t) {
                float p = __builtin_amdgcn_exp2f((s[t][r] - mnew) * LOG2E);
                s[t][r] = p;
                ps += p;
            }
#pragma unroll
            for (int msk = 1; msk <= 8; msk <<= 1)
                ps += __shfl_xor(ps, msk, 32);
            lrow[r] = lrow[r] * scale + ps;
            mrow[r] = mnew;
#pragma unroll
            for (int t = 0; t < 4; ++t) o[t][r] *= scale;
        }
        // ---- stage P (bf16, row-major 16x64) for C-layout -> A-layout transpose ----
#pragma unroll
        for (int t = 0; t < 4; ++t)
#pragma unroll
            for (int r = 0; r < 8; ++r)
                slab[(r + 8 * hi) * Dn + t * 16 + lo] = (bf16_t)s[t][r];
        asm volatile("s_wait_dscnt 0" ::: "memory");
        v16bf pf[2];
#pragma unroll
        for (int kc = 0; kc < 2; ++kc)
            pf[kc] = make_afrag(slab + lo * Dn + kc * 32 + hi * 16);
        asm volatile("" ::: "memory");
        // ---- O += P * V (V B-fragments contiguous thanks to XbT) ----
#pragma unroll
        for (int t = 0; t < 4; ++t) {
#pragma unroll
            for (int kc = 0; kc < 2; ++kc) {
                const bf16_t* vrow = XbTb + (long)(t * 16 + lo) * Nn + k0 + kc * 32 + hi * 16;
                v16bf vb = *(const v16bf*)vrow;
                o[t] = __builtin_amdgcn_wmma_f32_16x16x32_bf16(false, pf[kc], false, vb,
                                                               (short)0, o[t], false, false);
            }
        }
    }

    // ---- normalize, stage O_norm as bf16, transpose to A-layout ----
#pragma unroll
    for (int r = 0; r < 8; ++r) {
        float inv = 1.0f / lrow[r];
#pragma unroll
        for (int t = 0; t < 4; ++t)
            slab[(r + 8 * hi) * Dn + t * 16 + lo] = (bf16_t)(o[t][r] * inv);
    }
    asm volatile("s_wait_dscnt 0" ::: "memory");
    v16bf of[2];
#pragma unroll
    for (int kc = 0; kc < 2; ++kc)
        of[kc] = make_afrag(slab + lo * Dn + kc * 32 + hi * 16);

    // ---- projection: out = O_norm @ W (Wt is W transposed, rows contiguous) ----
    float* outp = Out + ((long)b * Nn + q0) * 64;
#pragma unroll
    for (int t = 0; t < 4; ++t) {
        v8f acc = {};
#pragma unroll
        for (int kc = 0; kc < 2; ++kc) {
            const bf16_t* wrow = Wt + (t * 16 + lo) * Dn + kc * 32 + hi * 16;
            v16bf wb = *(const v16bf*)wrow;
            acc = __builtin_amdgcn_wmma_f32_16x16x32_bf16(false, of[kc], false, wb,
                                                          (short)0, acc, false, false);
        }
#pragma unroll
        for (int r = 0; r < 8; ++r)
            outp[(long)(r + 8 * hi) * 64 + t * 16 + lo] = acc[r];
    }
}

extern "C" void kernel_launch(void* const* d_in, const int* in_sizes, int n_in,
                              void* d_out, int out_size, void* d_ws, size_t ws_size,
                              hipStream_t stream) {
    const float* X = (const float*)d_in[0];   // [B, N, D] fp32
    const float* W = (const float*)d_in[1];   // [D, 64] fp32
    char* ws = (char*)d_ws;
    bf16_t* Xb  = (bf16_t*)ws;                                   // 4 MiB
    bf16_t* XbT = (bf16_t*)(ws + (size_t)4 * 1024 * 1024);       // 4 MiB
    bf16_t* Wt  = (bf16_t*)(ws + (size_t)8 * 1024 * 1024);       // 8 KiB
    float* Out = (float*)d_out;

    int total = Bn * Nn * Dn;
    prep_kernel<<<(total + 255) / 256, 256, 0, stream>>>(X, W, Xb, XbT, Wt);

    dim3 grid(Nn / QT, Bn);
    flashattn_kernel<<<grid, 256, 0, stream>>>(Xb, XbT, Wt, Out);
}
